// GraphRes_28836410425487
// MI455X (gfx1250) — compile-verified, matrix operand order checked by reading, not run on hardware
//
#include <hip/hip_runtime.h>
#include <hip/hip_bf16.h>
#include <math.h>

// ---------------- problem constants ----------------
#define N_NODES   131072
#define N_EDGES   1048576
#define NUM_GRAPHS 16
#define NXG 22
#define NYG 22
#define C5N (NUM_GRAPHS * NXG * NYG)   // 7744 (divisible by 16)
#define P7  16
#define EPSB 1e-5f
#define VXc (16.0f/346.0f)
#define VYc (12.0f/260.0f)

typedef __attribute__((ext_vector_type(16))) _Float16 v16h;
typedef __attribute__((ext_vector_type(8)))  float    v8f;

union V16H { v16h v; uint4 q[2]; };

static __device__ __forceinline__ void atomicMaxFloat(float* addr, float val) {
  // works with -inf init; positive: signed-int order, negative: unsigned-min order
  if (val >= 0.f) atomicMax((int*)addr, __float_as_int(val));
  else            atomicMin((unsigned int*)addr, __float_as_uint(val));
}

// ---------------- small utility kernels ----------------
__global__ void fill_f32(float* __restrict__ p, float v, long n) {
  long i = (long)blockIdx.x * blockDim.x + threadIdx.x;
  if (i < n) p[i] = v;
}

// f32 [M,C] -> f16 [M,32] zero-padded (K padding for WMMA)
__global__ void pad_to_f16(const float* __restrict__ src, long M, int C,
                           _Float16* __restrict__ dst) {
  long i = (long)blockIdx.x * blockDim.x + threadIdx.x;
  if (i >= M * 32) return;
  long n = i >> 5; int c = (int)(i & 31);
  dst[i] = (c < C) ? (_Float16)src[n * C + c] : (_Float16)0.f;
}

// W [8,Cin,Cout] f32 -> WT f16 [8*Cout, 32(pad Cin)]:  WT[b*Cout+o][i] = W[b][i][o]
// (transposed so each lane's B fragment is two contiguous 16B chunks -> global_load_b128)
__global__ void pack_wT(const float* __restrict__ W, int Cin, int Cout,
                        _Float16* __restrict__ WT) {
  int t = blockIdx.x * blockDim.x + threadIdx.x;
  int Nc = 8 * Cout;
  if (t >= Nc * 32) return;
  int r = t >> 5, i = t & 31;           // r = b*Cout + o
  int b = r / Cout, o = r % Cout;
  WT[t] = (i < Cin) ? (_Float16)W[(b * Cin + i) * Cout + o] : (_Float16)0.f;
}

// open linear B-spline basis, kernel_size=2, dim=3: 8 trilinear weights per edge
__global__ void basis_kernel(const float* __restrict__ attr, float* __restrict__ basis, int E) {
  int e = blockIdx.x * blockDim.x + threadIdx.x;
  if (e >= E) return;
  float u0 = attr[e*3], u1 = attr[e*3+1], u2 = attr[e*3+2];
#pragma unroll
  for (int b = 0; b < 8; ++b)
    basis[(size_t)e*8 + b] = ((b&1)?u0:1.f-u0) * ((b&2)?u1:1.f-u1) * ((b&4)?u2:1.f-u2);
}

// ---------------- WMMA node GEMM:  Y[M, NT*16] = H[M,32] x W[32, NT*16] ----------------
// One wave per 16-row M-tile; A fragment loaded once (2x b128), then NT unrolled WMMAs
// over the N tiles, each with a 2x b128 B-fragment load from the transposed weights.
template<int NT>
__global__ void gemm_h16_wmma(const _Float16* __restrict__ H,
                              const _Float16* __restrict__ WT,  // [NT*16, 32]
                              _Float16* __restrict__ Y, int Mtiles) {
  const int Nc = NT * 16;
  int tileM = blockIdx.x * (blockDim.x >> 5) + (threadIdx.x >> 5);
  if (tileM >= Mtiles) return;                 // wave-uniform exit (EXEC all-ones for WMMA)
  int lane = threadIdx.x & 31;
  int hi   = lane >> 4;                        // 0: K {0..7,16..23}; 1: K {8..15,24..31}
  int l15  = lane & 15;

  // A fragment (ISA 7.12.2 16-bit A layout): row = tileM*16 + l15
  const uint4* arow = (const uint4*)(H + ((size_t)tileM * 16 + l15) * 32 + 8 * hi);
  V16H a; a.q[0] = arow[0]; a.q[1] = arow[2];  // +2 uint4 == +16 halves

#pragma unroll
  for (int t = 0; t < NT; ++t) {
    const uint4* bcol = (const uint4*)(WT + ((size_t)(t * 16 + l15)) * 32 + 8 * hi);
    V16H b; b.q[0] = bcol[0]; b.q[1] = bcol[2];
    v8f c = {};
    c = __builtin_amdgcn_wmma_f32_16x16x32_f16(false, a.v, false, b.v, (short)0, c,
                                               false, false);
    // D layout: VGPR r -> row (r + 8*hi), col = l15
#pragma unroll
    for (int r = 0; r < 8; ++r) {
      size_t row = (size_t)tileM * 16 + 8 * hi + r;
      Y[row * Nc + t * 16 + l15] = (_Float16)c[r];
    }
  }
}

// ---------------- dst-CSR build (edge list is static: built once, reused per phase) --------
__global__ void csr_count(const int* __restrict__ dst, const int* __restrict__ cluster,
                          const unsigned char* __restrict__ owner, int* __restrict__ indeg,
                          int E) {
  int e = blockIdx.x * blockDim.x + threadIdx.x;
  if (e >= E) return;
  if (owner && !owner[e]) return;
  int d = dst[e];
  if (cluster) d = cluster[d];
  atomicAdd(&indeg[d], 1);                      // integer: deterministic
}

// single-block exclusive scan (n up to ~10^6: chunked, deterministic)
__global__ void scan_exclusive(const int* __restrict__ cnt, int* __restrict__ offs, int n) {
  __shared__ int part[1024];
  int tid = threadIdx.x;
  int chunk = (n + blockDim.x - 1) / blockDim.x;
  int lo = tid * chunk, hi = lo + chunk; if (hi > n) hi = n;
  int s = 0;
  for (int i = lo; i < hi; ++i) s += cnt[i];
  part[tid] = s;
  __syncthreads();
  if (tid == 0) {
    int a = 0;
    for (int i = 0; i < (int)blockDim.x; ++i) { int t = part[i]; part[i] = a; a += t; }
  }
  __syncthreads();
  int a = part[tid];
  for (int i = lo; i < hi; ++i) { offs[i] = a; a += cnt[i]; }
}

__global__ void csr_scatter(const int* __restrict__ dst, const int* __restrict__ cluster,
                            const unsigned char* __restrict__ owner,
                            const int* __restrict__ offs, int* __restrict__ fill,
                            int* __restrict__ eids, int E) {
  int e = blockIdx.x * blockDim.x + threadIdx.x;
  if (e >= E) return;
  if (owner && !owner[e]) return;
  int d = dst[e];
  if (cluster) d = cluster[d];
  int p = offs[d] + atomicAdd(&fill[d], 1);
  eids[p] = e;
}

// ---------------- aggregation: one wave per dst node, lane = output channel --------------
// Replaces 33M f32 atomics/layer with gathers in fixed CSR order (no atomics at all).
// Per basis tap, lanes 0..COUT-1 read contiguous halves of y[src] -> coalesced 64B.
template<int COUT>
__global__ void agg_csr(const int* __restrict__ eids, const int* __restrict__ offs,
                        const int* __restrict__ indeg, const int* __restrict__ src,
                        const int* __restrict__ cluster, const float* __restrict__ basis,
                        const _Float16* __restrict__ Y,
                        float* __restrict__ agg, float* __restrict__ deg, int M) {
  int d = blockIdx.x * (blockDim.x >> 5) + (threadIdx.x >> 5);
  if (d >= M) return;
  int lane = threadIdx.x & 31;
  int o = lane % COUT;                          // lanes >= COUT mirror (broadcast loads)
  int beg = offs[d], n = indeg[d];
  float acc = 0.f;
  for (int i = 0; i < n; ++i) {
    int e = eids[beg + i];                      // wave-uniform -> scalar load
    int s = src[e];
    if (cluster) s = cluster[s];
    const float4* bq = (const float4*)(basis + (size_t)e * 8);  // wave-uniform, 32B
    float4 b0 = bq[0], b1 = bq[1];
    float bas[8] = {b0.x, b0.y, b0.z, b0.w, b1.x, b1.y, b1.z, b1.w};
    const _Float16* yr = Y + (size_t)s * (8 * COUT) + o;
    float m = 0.f;
#pragma unroll
    for (int b = 0; b < 8; ++b) m += bas[b] * (float)yr[b * COUT];
    acc += m;
  }
  if (lane < COUT) agg[(size_t)d * COUT + lane] = acc;
  if (lane == 0)  deg[d] = (float)n;
}

// ---------------- epilogue: mean-agg + ELU + (masked) BN statistics (deterministic) ----------
__global__ void ep1_stats(float* __restrict__ v, const float* __restrict__ deg,
                          const float* __restrict__ cnt, int M, int C,
                          float* __restrict__ partials) {
  const int tid = threadIdx.x;
  const int cpt = tid % C;              // channel
  const int stripe = tid / C;
  const int S = blockDim.x / C;         // C in {8,16,32} -> S power of two
  const int P = 2 * C + 1;
  long chunk = ((long)M + gridDim.x - 1) / gridDim.x;
  long n0 = (long)blockIdx.x * chunk;
  long n1 = n0 + chunk; if (n1 > M) n1 = M;
  float s = 0.f, sq = 0.f, cl = 0.f;
  for (long n = n0 + stripe; n < n1; n += S) {
    float m = cnt ? (cnt[n] > 0.f ? 1.f : 0.f) : 1.f;
    float x = v[n * C + cpt] / fmaxf(deg[n], 1.f);
    x = (x > 0.f) ? x : expm1f(x);      // ELU
    v[n * C + cpt] = x;
    s += x * m; sq += x * x * m;
    if (cpt == 0) cl += m;
  }
  __shared__ float red[256];
  red[tid] = s; __syncthreads();
  for (int off = S >> 1; off > 0; off >>= 1) {
    if (stripe < off) red[tid] += red[tid + off * C];
    __syncthreads();
  }
  if (stripe == 0) partials[blockIdx.x * P + cpt] = red[cpt];
  __syncthreads();
  red[tid] = sq; __syncthreads();
  for (int off = S >> 1; off > 0; off >>= 1) {
    if (stripe < off) red[tid] += red[tid + off * C];
    __syncthreads();
  }
  if (stripe == 0) partials[blockIdx.x * P + C + cpt] = red[cpt];
  __syncthreads();
  red[tid] = (cpt == 0) ? cl : 0.f; __syncthreads();
  for (int off = S >> 1; off > 0; off >>= 1) {
    if (stripe < off) red[tid] += red[tid + off * C];
    __syncthreads();
  }
  if (tid == 0) partials[blockIdx.x * P + 2 * C] = red[0];
}

__global__ void ep2_finalize(const float* __restrict__ partials, int NB, int C,
                             float* __restrict__ stats) {
  int c = threadIdx.x;
  if (c >= C) return;
  const int P = 2 * C + 1;
  float s = 0.f, sq = 0.f, n = 0.f;
  for (int p = 0; p < NB; ++p) {        // fixed order -> deterministic
    s  += partials[p * P + c];
    sq += partials[p * P + C + c];
    n  += partials[p * P + 2 * C];
  }
  n = fmaxf(n, 1.f);
  float mean = s / n;
  float var  = fmaxf(sq / n - mean * mean, 0.f);
  stats[c] = mean; stats[C + c] = var;
}

__global__ void ep3_apply(const float* __restrict__ v, const float* __restrict__ stats,
                          const float* __restrict__ gamma, const float* __restrict__ beta,
                          const float* __restrict__ sc, float* __restrict__ hout,
                          _Float16* __restrict__ h16, long M, int C) {
  long i = (long)blockIdx.x * blockDim.x + threadIdx.x;
  if (i >= M * C) return;
  int c = (int)(i % C);
  long n = i / C;
  float y = gamma[c] * (v[i] - stats[c]) * rsqrtf(stats[C + c] + EPSB) + beta[c];
  if (sc) y += sc[i];
  hout[i] = y;
  if (h16) h16[n * 32 + c] = (_Float16)y;  // pad cols stay zero from initial pad_to_f16
}

// ---------------- pool5 (voxel grid max-pool) ----------------
__global__ void pool5_assign(const float* __restrict__ pos, const int* __restrict__ batch,
                             const float* __restrict__ h, int* __restrict__ cluster,
                             float* __restrict__ cnt, float* __restrict__ possum,
                             float* __restrict__ hp) {
  int n = blockIdx.x * blockDim.x + threadIdx.x;
  if (n >= N_NODES) return;
  int ix = (int)floorf(pos[n*3+0] / VXc); ix = ix < 0 ? 0 : (ix > NXG-1 ? NXG-1 : ix);
  int iy = (int)floorf(pos[n*3+1] / VYc); iy = iy < 0 ? 0 : (iy > NYG-1 ? NYG-1 : iy);
  int cl = batch[n] * (NXG * NYG) + iy * NXG + ix;
  cluster[n] = cl;
  atomicAdd(&cnt[cl], 1.f);
  for (int k = 0; k < 3; ++k) atomicAdd(&possum[cl*3 + k], pos[n*3 + k]);
  for (int c = 0; c < 32; ++c) atomicMaxFloat(&hp[(size_t)cl*32 + c], h[(size_t)n*32 + c]);
}

__global__ void pool5_fin(const float* __restrict__ cnt, const float* __restrict__ possum,
                          float* __restrict__ posp, float* __restrict__ hp) {
  int p = blockIdx.x * blockDim.x + threadIdx.x;
  if (p >= C5N) return;
  float c = cnt[p];
  float inv = 1.f / fmaxf(c, 1.f);
  for (int k = 0; k < 3; ++k) posp[p*3 + k] = possum[p*3 + k] * inv;
  if (c <= 0.f) for (int ch = 0; ch < 32; ++ch) hp[(size_t)p*32 + ch] = 0.f;
}

// dedup coarse edges by first-setter ownership on a (cs,cd) bitmap (set semantics == sort+dedup)
__global__ void coarse_edges(const int* __restrict__ src, const int* __restrict__ dst,
                             const int* __restrict__ cluster, unsigned int* __restrict__ bitmap,
                             unsigned char* __restrict__ owner, int E) {
  int e = blockIdx.x * blockDim.x + threadIdx.x;
  if (e >= E) return;
  int cs = cluster[src[e]], cd = cluster[dst[e]];
  unsigned char w = 0;
  if (cs != cd) {
    unsigned int key = (unsigned int)cs * (unsigned int)C5N + (unsigned int)cd;
    unsigned int old = atomicOr(&bitmap[key >> 5], 1u << (key & 31));
    w = ((old >> (key & 31)) & 1u) ? 0 : 1;
  }
  owner[e] = w;
}

__global__ void mx_kernel(const int* __restrict__ src, const int* __restrict__ dst,
                          const int* __restrict__ cluster, const unsigned char* __restrict__ owner,
                          const float* __restrict__ posp, float* __restrict__ mx, int E) {
  int e = blockIdx.x * blockDim.x + threadIdx.x;
  if (e >= E || !owner[e]) return;
  int cs = cluster[src[e]], cd = cluster[dst[e]];
  for (int k = 0; k < 3; ++k) {
    float r = fabsf(posp[cd*3 + k] - posp[cs*3 + k]);   // >= 0, uint order == float order
    atomicMax((unsigned int*)mx, __float_as_uint(r));
  }
}

__global__ void basisc_kernel(const int* __restrict__ src, const int* __restrict__ dst,
                              const int* __restrict__ cluster, const unsigned char* __restrict__ owner,
                              const float* __restrict__ posp, const float* __restrict__ mx,
                              float* __restrict__ basis, int E) {
  int e = blockIdx.x * blockDim.x + threadIdx.x;
  if (e >= E || !owner[e]) return;
  int cs = cluster[src[e]], cd = cluster[dst[e]];
  float denom = 2.f * (*mx) + 1e-12f;
  float u[3];
  for (int k = 0; k < 3; ++k) {
    float a = (posp[cd*3 + k] - posp[cs*3 + k]) / denom + 0.5f;
    u[k] = fminf(fmaxf(a, 0.f), 1.f);
  }
#pragma unroll
  for (int b = 0; b < 8; ++b)
    basis[(size_t)e*8 + b] = ((b&1)?u[0]:1.f-u[0]) * ((b&2)?u[1]:1.f-u[1]) * ((b&4)?u[2]:1.f-u[2]);
}

// ---------------- pool7 (MaxPoolingX 4x4 per graph) + FC ----------------
__global__ void pool7_kernel(const float* __restrict__ posp, const float* __restrict__ cnt,
                             const float* __restrict__ hp, float* __restrict__ xo) {
  int p = blockIdx.x * blockDim.x + threadIdx.x;
  if (p >= C5N || cnt[p] <= 0.f) return;    // invalid clusters routed to dummy in ref
  int jx = (int)floorf(posp[p*3+0] / 0.25f); jx = jx < 0 ? 0 : (jx > 3 ? 3 : jx);
  int jy = (int)floorf(posp[p*3+1] / 0.25f); jy = jy < 0 ? 0 : (jy > 3 ? 3 : jy);
  int c7 = (p / (NXG * NYG)) * P7 + jy * 4 + jx;
  for (int c = 0; c < 32; ++c) atomicMaxFloat(&xo[c7*32 + c], hp[(size_t)p*32 + c]);
}

__global__ void fc_kernel(const float* __restrict__ xo, const float* __restrict__ fcw,
                          float* __restrict__ out) {
  int t = threadIdx.x;
  if (t >= 32) return;
  int g = t >> 1, j = t & 1;
  float acc = 0.f;
  for (int k = 0; k < P7 * 32; ++k) {
    float v = xo[g * (P7 * 32) + k];
    if (v == -INFINITY) v = 0.f;            // empty voxels -> 0 (ref: isneginf)
    acc += v * fcw[k * 2 + j];
  }
  out[g * 2 + j] = acc;
}

// ---------------- host-side dispatch helpers ----------------
static void launch_gemm(const _Float16* H, const _Float16* WT, _Float16* Y,
                        int M, int Nc, hipStream_t s) {
  int Mtiles = M / 16;
  dim3 grid((Mtiles + 7) / 8);
  switch (Nc) {
    case 64:  gemm_h16_wmma<4 ><<<grid, 256, 0, s>>>(H, WT, Y, Mtiles); break;
    case 128: gemm_h16_wmma<8 ><<<grid, 256, 0, s>>>(H, WT, Y, Mtiles); break;
    default:  gemm_h16_wmma<16><<<grid, 256, 0, s>>>(H, WT, Y, Mtiles); break;
  }
}

static void launch_agg(int Cout, const int* eids, const int* offs, const int* indeg,
                       const int* src, const int* cluster, const float* basis,
                       const _Float16* Y, float* agg, float* deg, int M, hipStream_t s) {
  dim3 grid((M + 7) / 8);                 // 8 waves / 256-thread block, one wave per dst
  switch (Cout) {
    case 8:  agg_csr<8 ><<<grid,256,0,s>>>(eids,offs,indeg,src,cluster,basis,Y,agg,deg,M); break;
    case 16: agg_csr<16><<<grid,256,0,s>>>(eids,offs,indeg,src,cluster,basis,Y,agg,deg,M); break;
    default: agg_csr<32><<<grid,256,0,s>>>(eids,offs,indeg,src,cluster,basis,Y,agg,deg,M); break;
  }
}

extern "C" void kernel_launch(void* const* d_in, const int* in_sizes, int n_in,
                              void* d_out, int out_size, void* d_ws, size_t ws_size,
                              hipStream_t stream) {
  (void)in_sizes; (void)n_in; (void)out_size; (void)ws_size;
  const float* x     = (const float*)d_in[0];
  const float* pos   = (const float*)d_in[1];
  const int*   batch = (const int*)d_in[2];
  const int*   srcI  = (const int*)d_in[3];
  const int*   dstI  = srcI + N_EDGES;
  const float* eattr = (const float*)d_in[4];
  const float* Ws[7], *gam[7], *bet[7];
  for (int i = 0; i < 7; ++i) { Ws[i]  = (const float*)d_in[5 + i];
                                gam[i] = (const float*)d_in[12 + i];
                                bet[i] = (const float*)d_in[19 + i]; }
  const float* fcw = (const float*)d_in[26];
  float* out = (float*)d_out;

  char* ws = (char*)d_ws;
  size_t off = 0;
  auto alloc = [&](size_t b) { size_t o = off; off += (b + 255) & ~(size_t)255; return o; };
  _Float16* y16   = (_Float16*)(ws + alloc((size_t)N_NODES * 256 * 2)); // 64 MB (L2-resident)
  float* basis    = (float*)(ws + alloc((size_t)N_EDGES * 8 * 4));      // 32 MB, reused x5 then coarse
  float* h        = (float*)(ws + alloc((size_t)N_NODES * 32 * 4));
  float* scb      = (float*)(ws + alloc((size_t)N_NODES * 16 * 4));
  float* agg      = (float*)(ws + alloc((size_t)N_NODES * 32 * 4));     // also holds v in-place
  _Float16* h16   = (_Float16*)(ws + alloc((size_t)N_NODES * 32 * 2));
  float* deg      = (float*)(ws + alloc((size_t)N_NODES * 4));
  _Float16* W16   = (_Float16*)(ws + alloc(32 * 256 * 2));
  float* partials = (float*)(ws + alloc(256 * 65 * 4));
  float* stats    = (float*)(ws + alloc(64 * 4));
  int*   cluster  = (int*)(ws + alloc((size_t)N_NODES * 4));
  float* cnt      = (float*)(ws + alloc((size_t)C5N * 4));
  float* possum   = (float*)(ws + alloc((size_t)C5N * 3 * 4));
  float* posp     = (float*)(ws + alloc((size_t)C5N * 3 * 4));
  float* hp       = (float*)(ws + alloc((size_t)C5N * 32 * 4));
  float* hpsc     = (float*)(ws + alloc((size_t)C5N * 32 * 4));
  unsigned char* owner = (unsigned char*)(ws + alloc(N_EDGES));
  unsigned int* bitmap = (unsigned int*)(ws + alloc((((size_t)C5N * C5N + 31) / 32) * 4));
  float* mx       = (float*)(ws + alloc(256));
  float* xo       = (float*)(ws + alloc((size_t)NUM_GRAPHS * P7 * 32 * 4));
  int* indeg      = (int*)(ws + alloc((size_t)N_NODES * 4));   // CSR (fine, then coarse reuse)
  int* offs       = (int*)(ws + alloc((size_t)N_NODES * 4));
  int* fillc      = (int*)(ws + alloc((size_t)N_NODES * 4));
  int* eids       = (int*)(ws + alloc((size_t)N_EDGES * 4));

  const int CH[8] = {1, 8, 16, 16, 16, 32, 32, 32};
  const int NB = 256;
  const int EB = (N_EDGES + 255) / 256;

  // spline basis once (edge_attr is fixed across layers 0..4)
  basis_kernel<<<EB, 256, 0, stream>>>(eattr, basis, N_EDGES);
  // initial features, zero-padded to K=32 for WMMA
  pad_to_f16<<<((size_t)N_NODES * 32 + 255) / 256, 256, 0, stream>>>(x, N_NODES, 1, h16);

  // dst-CSR for the fine graph (built once, reused by all 5 layers)
  hipMemsetAsync(indeg, 0, (size_t)N_NODES * 4, stream);
  csr_count<<<EB, 256, 0, stream>>>(dstI, nullptr, nullptr, indeg, N_EDGES);
  scan_exclusive<<<1, 1024, 0, stream>>>(indeg, offs, N_NODES);
  hipMemsetAsync(fillc, 0, (size_t)N_NODES * 4, stream);
  csr_scatter<<<EB, 256, 0, stream>>>(dstI, nullptr, nullptr, offs, fillc, eids, N_EDGES);

  // ---- fine-graph layers 0..4 ----
  for (int l = 0; l < 5; ++l) {
    int Cin = CH[l], Cout = CH[l + 1], Nc = 8 * Cout;
    pack_wT<<<(Nc * 32 + 255) / 256, 256, 0, stream>>>(Ws[l], Cin, Cout, W16);
    launch_gemm(h16, W16, y16, N_NODES, Nc, stream);
    launch_agg(Cout, eids, offs, indeg, srcI, nullptr, basis, y16, agg, deg, N_NODES, stream);
    ep1_stats<<<NB, 256, 0, stream>>>(agg, deg, nullptr, N_NODES, Cout, partials);
    ep2_finalize<<<1, 32, 0, stream>>>(partials, NB, Cout, stats);
    const float* scp = (l == 3) ? scb : nullptr;       // h = block3(...) + sc
    _Float16* h16o   = (l < 4) ? h16 : nullptr;
    long tot = (long)N_NODES * Cout;
    ep3_apply<<<(tot + 255) / 256, 256, 0, stream>>>(agg, stats, gam[l], bet[l], scp, h, h16o,
                                                     N_NODES, Cout);
    if (l == 1) hipMemcpyAsync(scb, h, (size_t)N_NODES * 16 * 4,
                               hipMemcpyDeviceToDevice, stream);  // sc = h after block1
  }

  // ---- pool5 ----
  fill_f32<<<((size_t)C5N * 32 + 255) / 256, 256, 0, stream>>>(hp, -INFINITY, (long)C5N * 32);
  hipMemsetAsync(cnt, 0, (size_t)C5N * 4, stream);
  hipMemsetAsync(possum, 0, (size_t)C5N * 3 * 4, stream);
  pool5_assign<<<(N_NODES + 255) / 256, 256, 0, stream>>>(pos, batch, h, cluster, cnt, possum, hp);
  pool5_fin<<<(C5N + 255) / 256, 256, 0, stream>>>(cnt, possum, posp, hp);
  hipMemcpyAsync(hpsc, hp, (size_t)C5N * 32 * 4, hipMemcpyDeviceToDevice, stream); // sc = hp
  hipMemsetAsync(bitmap, 0, (((size_t)C5N * C5N + 31) / 32) * 4, stream);
  coarse_edges<<<EB, 256, 0, stream>>>(srcI, dstI, cluster, bitmap, owner, N_EDGES);
  hipMemsetAsync(mx, 0, 4, stream);
  mx_kernel<<<EB, 256, 0, stream>>>(srcI, dstI, cluster, owner, posp, mx, N_EDGES);
  basisc_kernel<<<EB, 256, 0, stream>>>(srcI, dstI, cluster, owner, posp, mx, basis, N_EDGES);

  // dst-CSR for the coarse graph (owner edges, cluster buckets; reuses the CSR arrays)
  hipMemsetAsync(indeg, 0, (size_t)C5N * 4, stream);
  csr_count<<<EB, 256, 0, stream>>>(dstI, cluster, owner, indeg, N_EDGES);
  scan_exclusive<<<1, 1024, 0, stream>>>(indeg, offs, C5N);
  hipMemsetAsync(fillc, 0, (size_t)C5N * 4, stream);
  csr_scatter<<<EB, 256, 0, stream>>>(dstI, cluster, owner, offs, fillc, eids, N_EDGES);

  // ---- coarse-graph layers 5..6 ----
  pad_to_f16<<<((size_t)C5N * 32 + 255) / 256, 256, 0, stream>>>(hp, C5N, 32, h16);
  for (int l = 5; l < 7; ++l) {
    const int Nc = 256;
    pack_wT<<<(Nc * 32 + 255) / 256, 256, 0, stream>>>(Ws[l], 32, 32, W16);
    launch_gemm(h16, W16, y16, C5N, Nc, stream);
    launch_agg(32, eids, offs, indeg, srcI, cluster, basis, y16, agg, deg, C5N, stream);
    ep1_stats<<<NB, 256, 0, stream>>>(agg, deg, cnt, C5N, 32, partials);   // masked BN
    ep2_finalize<<<1, 32, 0, stream>>>(partials, NB, 32, stats);
    const float* scp = (l == 6) ? hpsc : nullptr;      // hp = block6(...) + sc
    _Float16* h16o   = (l == 5) ? h16 : nullptr;
    long tot = (long)C5N * 32;
    ep3_apply<<<(tot + 255) / 256, 256, 0, stream>>>(agg, stats, gam[l], bet[l], scp, hp, h16o,
                                                     C5N, 32);
  }

  // ---- pool7 + FC ----
  fill_f32<<<((size_t)NUM_GRAPHS * P7 * 32 + 255) / 256, 256, 0, stream>>>(
      xo, -INFINITY, (long)NUM_GRAPHS * P7 * 32);
  pool7_kernel<<<(C5N + 255) / 256, 256, 0, stream>>>(posp, cnt, hp, xo);
  fc_kernel<<<1, 32, 0, stream>>>(xo, fcw, out);
}
